// HDBlock_61521111548221
// MI455X (gfx1250) — compile-verified, matrix operand order checked by reading, not run on hardware
//
#include <hip/hip_runtime.h>
#include <cstdint>
#include <cstddef>

// ---------------- problem constants ----------------
constexpr int DIM   = 2560;
constexpr int HEADS = 20;
constexpr int HD    = 128;
constexpr int NE    = 4;      // experts
constexpr int HE    = 6912;   // expert hidden
constexpr int HS    = 3584;   // shared hidden
constexpr int BATCH = 2;
constexpr int SEQ   = 2048;
constexpr int MROWS = BATCH * SEQ;        // 4096
constexpr int D6    = 6 * DIM;            // 15360

// ---------------- WMMA types ----------------
typedef __attribute__((ext_vector_type(16))) __bf16 bf16x16;
typedef __attribute__((ext_vector_type(2)))  __bf16 bf16x2;
typedef __attribute__((ext_vector_type(8)))  float  f32x8;
typedef __attribute__((ext_vector_type(2)))  float  f32x2;

union FragBF { bf16x16 v; unsigned int u[8]; };
union FragF  { f32x8   v; float        f[8]; };

__device__ __forceinline__ f32x8 wmma_bf16(bf16x16 a, bf16x16 b, f32x8 c) {
    // v_wmma_f32_16x16x32_bf16  (neg_a, A, neg_b, B, c_mod, C, reuse_a, reuse_b)
    return __builtin_amdgcn_wmma_f32_16x16x32_bf16(false, a, false, b, (short)0, c, false, false);
}

// hardware bf16 conversion paths (v_cvt_pk_bf16_f32 / v_cvt_bf16_f32)
__device__ __forceinline__ unsigned int pack2(float lo, float hi) {
    f32x2 f; f.x = lo; f.y = hi;
    bf16x2 b = __builtin_convertvector(f, bf16x2);
    union { bf16x2 b; unsigned int u; } c; c.b = b;
    return c.u;
}
__device__ __forceinline__ unsigned short f2bf(float f) {
    union { __bf16 b; unsigned short u; } c; c.b = (__bf16)f;
    return c.u;
}
// 16-bit A/B fragment K offset for vgpr j (ISA 7.12.2): pairs (2j,2j+1),
// vgprs 4..7 -> +16, upper lane half -> +8.
__device__ __forceinline__ int koff(int lane, int j) {
    return ((j < 4) ? (2 * j) : (16 + 2 * (j - 4))) + ((lane >> 4) << 3);
}
// fast reciprocal (single v_rcp_f32, no Newton refinement / div fixup)
__device__ __forceinline__ float frcp(float x) { return __builtin_amdgcn_rcpf(x); }
__device__ __forceinline__ float siluf(float x) { return x * frcp(1.0f + __expf(-x)); }

// =====================================================================
// Generic tiled GEMM: C[M,N] (= or +=) rowScale * (A[M,K] @ B[K,N]), fp32
// in memory, bf16 WMMA internally. BM=BN=128, BK=32, 256 threads / 8 waves,
// each wave owns a 64x32 slab of C (4x2 tiles of 16x16).
// Template-specialized epilogue: no branches, per-row scale hoisted.
// =====================================================================
template <bool SCALED, bool ACC>
__global__ void __launch_bounds__(256)
hd455_gemm(const float* __restrict__ A, const float* __restrict__ Bm,
           float* __restrict__ C, int M, int N, int K,
           const float* __restrict__ rowScale, int rsStride)
{
    __shared__ unsigned short As[128 * 32];
    __shared__ unsigned short Bs[128 * 32];   // stored [n][k]: K-pairs contiguous

    const int t = threadIdx.x, lane = t & 31, wid = t >> 5;
    const int rowBase = blockIdx.y * 128, colBase = blockIdx.x * 128;
    const int wr = wid >> 2, wc = wid & 3;    // 2x4 wave grid

    FragF acc[4][2];
#pragma unroll
    for (int a = 0; a < 4; ++a)
#pragma unroll
        for (int b = 0; b < 2; ++b)
#pragma unroll
            for (int i = 0; i < 8; ++i) acc[a][b].f[i] = 0.f;

    for (int k0 = 0; k0 < K; k0 += 32) {
        __syncthreads();
        // A tile 128x32: float4 loads, packed b64 LDS stores
#pragma unroll
        for (int i = 0; i < 4; ++i) {
            int idx = t + i * 256;            // 0..1023 float4 units
            int r = idx >> 3, c4 = (idx & 7) * 4;
            float4 a = *(const float4*)&A[(size_t)(rowBase + r) * K + k0 + c4];
            uint2 p; p.x = pack2(a.x, a.y); p.y = pack2(a.z, a.w);
            *(uint2*)&As[r * 32 + c4] = p;
        }
        // B tile 32x128 -> Bs[n][k]: two adjacent K rows per unit so pairs
        // pack in registers and LDS stores are b32.
#pragma unroll
        for (int i = 0; i < 2; ++i) {
            int u = t + i * 256;              // 0..511 units of (2k x 4n)
            int kk = (u >> 5) * 2, n4 = (u & 31) * 4;
            const float* bp = &Bm[(size_t)(k0 + kk) * N + colBase + n4];
            float4 b0 = *(const float4*)bp;
            float4 b1 = *(const float4*)(bp + N);
            *(unsigned int*)&Bs[(n4 + 0) * 32 + kk] = pack2(b0.x, b1.x);
            *(unsigned int*)&Bs[(n4 + 1) * 32 + kk] = pack2(b0.y, b1.y);
            *(unsigned int*)&Bs[(n4 + 2) * 32 + kk] = pack2(b0.z, b1.z);
            *(unsigned int*)&Bs[(n4 + 3) * 32 + kk] = pack2(b0.w, b1.w);
        }
        __syncthreads();

        FragBF af[4], bf[2];
#pragma unroll
        for (int rt = 0; rt < 4; ++rt) {
            int r = wr * 64 + rt * 16 + (lane & 15);
#pragma unroll
            for (int j = 0; j < 8; ++j)
                af[rt].u[j] = *(const unsigned int*)&As[r * 32 + koff(lane, j)];
        }
#pragma unroll
        for (int ct = 0; ct < 2; ++ct) {
            int n = wc * 32 + ct * 16 + (lane & 15);
#pragma unroll
            for (int j = 0; j < 8; ++j)
                bf[ct].u[j] = *(const unsigned int*)&Bs[n * 32 + koff(lane, j)];
        }
#pragma unroll
        for (int rt = 0; rt < 4; ++rt)
#pragma unroll
            for (int ct = 0; ct < 2; ++ct)
                acc[rt][ct].v = wmma_bf16(af[rt].v, bf[ct].v, acc[rt][ct].v);
    }

    const int half8 = (lane >> 4) << 3;
#pragma unroll
    for (int rt = 0; rt < 4; ++rt)
#pragma unroll
        for (int r = 0; r < 8; ++r) {
            int row = rowBase + wr * 64 + rt * 16 + r + half8;
            float sc = SCALED ? rowScale[(size_t)row * rsStride] : 1.0f;
#pragma unroll
            for (int ct = 0; ct < 2; ++ct) {
                int col = colBase + wc * 32 + ct * 16 + (lane & 15);
                size_t idx = (size_t)row * N + col;
                float val = SCALED ? (sc * acc[rt][ct].f[r]) : acc[rt][ct].f[r];
                if (ACC) C[idx] += val; else C[idx] = val;
            }
        }
}

// =====================================================================
// Dual GEMM with SwiGLU epilogue: Out = silu(A@W1) * (A@W3)
// BM=64, BN=128, BK=32; wave owns 32x32 (2x2 tiles) per accumulator set.
// =====================================================================
__global__ void __launch_bounds__(256)
hd455_gemm_dual_silu(const float* __restrict__ A, const float* __restrict__ W1,
                     const float* __restrict__ W3, float* __restrict__ Out,
                     int M, int N, int K)
{
    __shared__ unsigned short As[64 * 32];
    __shared__ unsigned short B1s[128 * 32];
    __shared__ unsigned short B3s[128 * 32];

    const int t = threadIdx.x, lane = t & 31, wid = t >> 5;
    const int rowBase = blockIdx.y * 64, colBase = blockIdx.x * 128;
    const int wr = wid >> 2, wc = wid & 3;

    FragF acc1[2][2], acc3[2][2];
#pragma unroll
    for (int a = 0; a < 2; ++a)
#pragma unroll
        for (int b = 0; b < 2; ++b)
#pragma unroll
            for (int i = 0; i < 8; ++i) { acc1[a][b].f[i] = 0.f; acc3[a][b].f[i] = 0.f; }

    for (int k0 = 0; k0 < K; k0 += 32) {
        __syncthreads();
#pragma unroll
        for (int i = 0; i < 2; ++i) {         // A tile 64x32: 512 float4 units
            int idx = t + i * 256;
            int r = idx >> 3, c4 = (idx & 7) * 4;
            float4 a = *(const float4*)&A[(size_t)(rowBase + r) * K + k0 + c4];
            uint2 p; p.x = pack2(a.x, a.y); p.y = pack2(a.z, a.w);
            *(uint2*)&As[r * 32 + c4] = p;
        }
#pragma unroll
        for (int i = 0; i < 2; ++i) {         // B tiles: units of (2k x 4n)
            int u = t + i * 256;
            int kk = (u >> 5) * 2, n4 = (u & 31) * 4;
            size_t g = (size_t)(k0 + kk) * N + colBase + n4;
            float4 a0 = *(const float4*)&W1[g];
            float4 a1 = *(const float4*)&W1[g + N];
            *(unsigned int*)&B1s[(n4 + 0) * 32 + kk] = pack2(a0.x, a1.x);
            *(unsigned int*)&B1s[(n4 + 1) * 32 + kk] = pack2(a0.y, a1.y);
            *(unsigned int*)&B1s[(n4 + 2) * 32 + kk] = pack2(a0.z, a1.z);
            *(unsigned int*)&B1s[(n4 + 3) * 32 + kk] = pack2(a0.w, a1.w);
            float4 c0 = *(const float4*)&W3[g];
            float4 c1 = *(const float4*)&W3[g + N];
            *(unsigned int*)&B3s[(n4 + 0) * 32 + kk] = pack2(c0.x, c1.x);
            *(unsigned int*)&B3s[(n4 + 1) * 32 + kk] = pack2(c0.y, c1.y);
            *(unsigned int*)&B3s[(n4 + 2) * 32 + kk] = pack2(c0.z, c1.z);
            *(unsigned int*)&B3s[(n4 + 3) * 32 + kk] = pack2(c0.w, c1.w);
        }
        __syncthreads();

        FragBF af[2], b1[2], b3[2];
#pragma unroll
        for (int rt = 0; rt < 2; ++rt) {
            int r = wr * 32 + rt * 16 + (lane & 15);
#pragma unroll
            for (int j = 0; j < 8; ++j)
                af[rt].u[j] = *(const unsigned int*)&As[r * 32 + koff(lane, j)];
        }
#pragma unroll
        for (int ct = 0; ct < 2; ++ct) {
            int n = wc * 32 + ct * 16 + (lane & 15);
#pragma unroll
            for (int j = 0; j < 8; ++j) {
                b1[ct].u[j] = *(const unsigned int*)&B1s[n * 32 + koff(lane, j)];
                b3[ct].u[j] = *(const unsigned int*)&B3s[n * 32 + koff(lane, j)];
            }
        }
#pragma unroll
        for (int rt = 0; rt < 2; ++rt)
#pragma unroll
            for (int ct = 0; ct < 2; ++ct) {
                acc1[rt][ct].v = wmma_bf16(af[rt].v, b1[ct].v, acc1[rt][ct].v);
                acc3[rt][ct].v = wmma_bf16(af[rt].v, b3[ct].v, acc3[rt][ct].v);
            }
    }

    const int half8 = (lane >> 4) << 3;
#pragma unroll
    for (int rt = 0; rt < 2; ++rt)
#pragma unroll
        for (int ct = 0; ct < 2; ++ct)
#pragma unroll
            for (int r = 0; r < 8; ++r) {
                int row = rowBase + wr * 32 + rt * 16 + r + half8;
                int col = colBase + wc * 32 + ct * 16 + (lane & 15);
                Out[(size_t)row * N + col] = siluf(acc1[rt][ct].f[r]) * acc3[rt][ct].f[r];
            }
}

// =====================================================================
// Flash attention. Grid: (SEQ/128, BATCH*HEADS). 8 waves/block, wave = 16 q rows.
// q,k,v: (B,S,DIM) fp32 with head h at column h*HD. Scale folded into Q.
// =====================================================================
__global__ void __launch_bounds__(256)
hd455_attn(const float* __restrict__ q, const float* __restrict__ k,
           const float* __restrict__ v, float* __restrict__ o)
{
    __shared__ unsigned short Ks[32 * 128];     // [key][d]  pairs contiguous along d
    __shared__ unsigned short Vs[128 * 32];     // [d][key]  pairs contiguous along key
    __shared__ unsigned short Pb[8][16 * 32];   // wave-private P staging

    const int t = threadIdx.x, lane = t & 31, wid = t >> 5;
    const int bh = blockIdx.y, b = bh / HEADS, h = bh % HEADS;
    const int q0 = blockIdx.x * 128 + wid * 16;
    const float scale = 0.08838834764831845f;   // 1/sqrt(128)

    // Q A-fragments, 4 chunks of K=32 over HD=128 (pairs contiguous -> float2)
    FragBF aQ[4];
    {
        size_t qb = ((size_t)(b * SEQ) + q0 + (lane & 15)) * DIM + (size_t)h * HD;
#pragma unroll
        for (int i = 0; i < 4; ++i)
#pragma unroll
            for (int j = 0; j < 8; ++j) {
                int d = i * 32 + koff(lane, j);
                float2 qq = *(const float2*)&q[qb + d];
                aQ[i].u[j] = pack2(qq.x * scale, qq.y * scale);
            }
    }

    FragF O[8];
#pragma unroll
    for (int c = 0; c < 8; ++c)
#pragma unroll
        for (int i = 0; i < 8; ++i) O[c].f[i] = 0.f;
    float mrow[8], lrow[8];
#pragma unroll
    for (int r = 0; r < 8; ++r) { mrow[r] = -1e30f; lrow[r] = 0.f; }

    for (int kb = 0; kb < SEQ; kb += 32) {
        __syncthreads();
        size_t kvb = ((size_t)(b * SEQ) + kb) * DIM + (size_t)h * HD;
        // K: 1024 float4 units -> packed b64 LDS stores
#pragma unroll
        for (int i = 0; i < 4; ++i) {
            int idx = t + i * 256;
            int key = idx >> 5, d4 = (idx & 31) * 4;
            float4 kv = *(const float4*)&k[kvb + (size_t)key * DIM + d4];
            uint2 p; p.x = pack2(kv.x, kv.y); p.y = pack2(kv.z, kv.w);
            *(uint2*)&Ks[key * 128 + d4] = p;
        }
        // V transposed: units of (2 keys x 4 d), b32 LDS stores
#pragma unroll
        for (int i = 0; i < 2; ++i) {
            int u = t + i * 256;
            int kk = (u >> 5) * 2, d4 = (u & 31) * 4;
            float4 v0 = *(const float4*)&v[kvb + (size_t)kk * DIM + d4];
            float4 v1 = *(const float4*)&v[kvb + (size_t)(kk + 1) * DIM + d4];
            *(unsigned int*)&Vs[(d4 + 0) * 32 + kk] = pack2(v0.x, v1.x);
            *(unsigned int*)&Vs[(d4 + 1) * 32 + kk] = pack2(v0.y, v1.y);
            *(unsigned int*)&Vs[(d4 + 2) * 32 + kk] = pack2(v0.z, v1.z);
            *(unsigned int*)&Vs[(d4 + 3) * 32 + kk] = pack2(v0.w, v1.w);
        }
        __syncthreads();

        // scores: two 16x16 tiles (keys kb..+15, kb+16..+31)
        FragF s0, s1;
#pragma unroll
        for (int i = 0; i < 8; ++i) { s0.f[i] = 0.f; s1.f[i] = 0.f; }
        const int colk = lane & 15;
#pragma unroll
        for (int c = 0; c < 4; ++c) {
            FragBF bK0, bK1;
#pragma unroll
            for (int j = 0; j < 8; ++j) {
                int dk = c * 32 + koff(lane, j);
                bK0.u[j] = *(const unsigned int*)&Ks[colk * 128 + dk];
                bK1.u[j] = *(const unsigned int*)&Ks[(colk + 16) * 128 + dk];
            }
            s0.v = wmma_bf16(aQ[c].v, bK0.v, s0.v);
            s1.v = wmma_bf16(aQ[c].v, bK1.v, s1.v);
        }

        // online softmax update per row (row = r + 8*(lane>=16))
#pragma unroll
        for (int r = 0; r < 8; ++r) {
            float mv = fmaxf(s0.f[r], s1.f[r]);
#pragma unroll
            for (int msk = 1; msk < 16; msk <<= 1) mv = fmaxf(mv, __shfl_xor(mv, msk, 32));
            float nm = fmaxf(mrow[r], mv);
            float alpha = __expf(mrow[r] - nm);
            float p0 = __expf(s0.f[r] - nm);
            float p1 = __expf(s1.f[r] - nm);
            float ps = p0 + p1;
#pragma unroll
            for (int msk = 1; msk < 16; msk <<= 1) ps += __shfl_xor(ps, msk, 32);
            lrow[r] = lrow[r] * alpha + ps;
            mrow[r] = nm;
            s0.f[r] = p0; s1.f[r] = p1;
#pragma unroll
            for (int c = 0; c < 8; ++c) O[c].f[r] *= alpha;
        }

        // P (16x32) -> LDS -> A-fragment layout
        unsigned short* P = &Pb[wid][0];
        const int half8 = (lane >> 4) << 3;
#pragma unroll
        for (int r = 0; r < 8; ++r) {
            int rr = r + half8;
            P[rr * 32 + (lane & 15)]      = f2bf(s0.f[r]);
            P[rr * 32 + 16 + (lane & 15)] = f2bf(s1.f[r]);
        }
        asm volatile("s_wait_dscnt 0" ::: "memory");
        FragBF aP;
#pragma unroll
        for (int j = 0; j < 8; ++j)
            aP.u[j] = *(const unsigned int*)&P[(lane & 15) * 32 + koff(lane, j)];

        // O += P @ V  (8 d-chunks of 16)
#pragma unroll
        for (int c = 0; c < 8; ++c) {
            FragBF bV;
#pragma unroll
            for (int j = 0; j < 8; ++j)
                bV.u[j] = *(const unsigned int*)&Vs[(c * 16 + (lane & 15)) * 32 + koff(lane, j)];
            O[c].v = wmma_bf16(aP.v, bV.v, O[c].v);
        }
    }

    // epilogue: multiply by rcp(l) (hoisted per row), store
    const int half8 = (lane >> 4) << 3;
    size_t ob = ((size_t)(b * SEQ) + q0) * DIM + (size_t)h * HD;
#pragma unroll
    for (int r = 0; r < 8; ++r) {
        int rr = r + half8;
        float inv = frcp(lrow[r]);
#pragma unroll
        for (int c = 0; c < 8; ++c)
            o[ob + (size_t)rr * DIM + c * 16 + (lane & 15)] = O[c].f[r] * inv;
    }
}

// =====================================================================
// Elementwise / small kernels
// =====================================================================
__global__ void hd455_silu_small(const float* __restrict__ in, float* __restrict__ out, int n) {
    int i = blockIdx.x * 256 + threadIdx.x;
    if (i < n) out[i] = siluf(in[i]);
}

// mod = silu(adaln) @ adaln_w + adaln_b   (M=2, K=2560, N=15360)
__global__ void hd455_adaln_gemm(const float* __restrict__ sa, const float* __restrict__ W,
                                 const float* __restrict__ bias, float* __restrict__ mod) {
    int idx = blockIdx.x * 256 + threadIdx.x;       // BATCH*D6
    int b = idx / D6, n = idx % D6;
    float acc = bias[n];
    for (int d = 0; d < DIM; ++d)
        acc += sa[b * DIM + d] * W[(size_t)d * D6 + n];
    mod[idx] = acc;
}

// out = LN(x) * (1 + sc) + sh  per row of DIM
__global__ void __launch_bounds__(256)
hd455_ln_mod(const float* __restrict__ x, const float* __restrict__ mod,
             int shOff, int scOff, float* __restrict__ out) {
    int row = blockIdx.x, b = row / SEQ, t = threadIdx.x;
    const float* xr = x + (size_t)row * DIM;
    __shared__ float r1[256], r2[256];
    float s = 0.f, s2 = 0.f;
    for (int d = t; d < DIM; d += 256) { float v = xr[d]; s += v; s2 += v * v; }
    r1[t] = s; r2[t] = s2; __syncthreads();
    for (int off = 128; off > 0; off >>= 1) {
        if (t < off) { r1[t] += r1[t + off]; r2[t] += r2[t + off]; }
        __syncthreads();
    }
    float mean = r1[0] / DIM;
    float var  = r2[0] / DIM - mean * mean;
    float rs = rsqrtf(var + 1e-6f);
    const float* sh = mod + (size_t)b * D6 + shOff;
    const float* sc = mod + (size_t)b * D6 + scOff;
    float* orow = out + (size_t)row * DIM;
    for (int d = t; d < DIM; d += 256)
        orow[d] = (xr[d] - mean) * rs * (1.0f + sc[d]) + sh[d];
}

// x = x * rsqrt(mean(x^2)+eps) * w  per row (in place)
__global__ void __launch_bounds__(256)
hd455_rms(float* __restrict__ x, const float* __restrict__ w) {
    int row = blockIdx.x, t = threadIdx.x;
    float* xr = x + (size_t)row * DIM;
    __shared__ float r1[256];
    float s2 = 0.f;
    for (int d = t; d < DIM; d += 256) { float v = xr[d]; s2 += v * v; }
    r1[t] = s2; __syncthreads();
    for (int off = 128; off > 0; off >>= 1) {
        if (t < off) r1[t] += r1[t + off];
        __syncthreads();
    }
    float sc = rsqrtf(r1[0] / DIM + 1e-6f);
    for (int d = t; d < DIM; d += 256) xr[d] = xr[d] * sc * w[d];
}

// interleaved RoPE on q and k, one thread per (b,s,h,pair)
__global__ void hd455_rope(float* __restrict__ q, float* __restrict__ k,
                           const float* __restrict__ rope) {
    const int HALF = HD / 2;
    int idx = blockIdx.x * 256 + threadIdx.x;       // B*S*HEADS*HALF
    int j  = idx % HALF;
    int hh = (idx / HALF) % HEADS;
    int s  = (idx / (HALF * HEADS)) % SEQ;
    int b  = idx / (HALF * HEADS * SEQ);
    const float4 r = *(const float4*)&rope[((size_t)s * HALF + j) * 4];
    size_t base = ((size_t)(b * SEQ + s)) * DIM + (size_t)hh * HD + 2 * j;
    float2 xq = *(const float2*)&q[base];
    float2 xk = *(const float2*)&k[base];
    float2 oq, ok;
    oq.x = r.x * xq.x + r.y * xq.y;  oq.y = r.z * xq.x + r.w * xq.y;
    ok.x = r.x * xk.x + r.y * xk.y;  ok.y = r.z * xk.x + r.w * xk.y;
    *(float2*)&q[base] = oq;
    *(float2*)&k[base] = ok;
}

// gating: softmax over 4 expert logits, keep top-2 scores, zero the rest
__global__ void __launch_bounds__(64)
hd455_gate(const float* __restrict__ h, const float* __restrict__ gw, float* __restrict__ cw) {
    int row = blockIdx.x, t = threadIdx.x;
    const float* hr = h + (size_t)row * DIM;
    __shared__ float red[NE][64];
    float p[NE] = {0.f, 0.f, 0.f, 0.f};
    for (int d = t; d < DIM; d += 64) {
        float v = hr[d];
#pragma unroll
        for (int e = 0; e < NE; ++e) p[e] += v * gw[e * DIM + d];
    }
#pragma unroll
    for (int e = 0; e < NE; ++e) red[e][t] = p[e];
    __syncthreads();
    for (int off = 32; off > 0; off >>= 1) {
        if (t < off)
#pragma unroll
            for (int e = 0; e < NE; ++e) red[e][t] += red[e][t + off];
        __syncthreads();
    }
    if (t == 0) {
        float lg[NE], sc[NE];
#pragma unroll
        for (int e = 0; e < NE; ++e) lg[e] = red[e][0];
        float mx = lg[0];
#pragma unroll
        for (int e = 1; e < NE; ++e) mx = fmaxf(mx, lg[e]);
        float sum = 0.f;
#pragma unroll
        for (int e = 0; e < NE; ++e) { sc[e] = __expf(lg[e] - mx); sum += sc[e]; }
        float inv = frcp(sum);
#pragma unroll
        for (int e = 0; e < NE; ++e) sc[e] *= inv;
        int i0 = 0;
#pragma unroll
        for (int e = 1; e < NE; ++e) if (sc[e] > sc[i0]) i0 = e;
        int i1 = -1;
#pragma unroll
        for (int e = 0; e < NE; ++e)
            if (e != i0 && (i1 < 0 || sc[e] > sc[i1])) i1 = e;
#pragma unroll
        for (int e = 0; e < NE; ++e)
            cw[(size_t)row * NE + e] = (e == i0 || e == i1) ? sc[e] : 0.f;
    }
}

// out = xin + g[b, d] * tin   (g = mod slice)
__global__ void hd455_residual(const float* __restrict__ xin, const float* __restrict__ tin,
                               const float* __restrict__ mod, int gOff,
                               float* __restrict__ out, int total) {
    int idx = blockIdx.x * 256 + threadIdx.x;
    if (idx >= total) return;
    int b = idx / (SEQ * DIM);
    int d = idx % DIM;
    out[idx] = xin[idx] + mod[(size_t)b * D6 + gOff + d] * tin[idx];
}

// =====================================================================
// Orchestration
// =====================================================================
extern "C" void kernel_launch(void* const* d_in, const int* in_sizes, int n_in,
                              void* d_out, int out_size, void* d_ws, size_t ws_size,
                              hipStream_t stream) {
    (void)in_sizes; (void)n_in; (void)out_size; (void)ws_size;
    const float* x_in  = (const float*)d_in[0];
    const float* adaln = (const float*)d_in[1];
    const float* rope  = (const float*)d_in[2];
    const float* adaw  = (const float*)d_in[3];
    const float* adab  = (const float*)d_in[4];
    const float* wq    = (const float*)d_in[5];
    const float* wk    = (const float*)d_in[6];
    const float* wv    = (const float*)d_in[7];
    const float* qrms  = (const float*)d_in[8];
    const float* krms  = (const float*)d_in[9];
    const float* wo    = (const float*)d_in[10];
    const float* gw    = (const float*)d_in[11];
    const float* ew1   = (const float*)d_in[12];
    const float* ew3   = (const float*)d_in[13];
    const float* ew2   = (const float*)d_in[14];
    const float* sw1   = (const float*)d_in[15];
    const float* sw3   = (const float*)d_in[16];
    const float* sw2   = (const float*)d_in[17];
    float* out = (float*)d_out;
    float* ws  = (float*)d_ws;

    const size_t NBS = (size_t)MROWS * DIM;          // 10,485,760 floats
    float* sa   = ws;                                 // BATCH*DIM
    float* mod  = ws + 8192;                          // BATCH*6*DIM
    float* cw   = ws + 40960;                         // MROWS*NE
    float* h    = ws + 65536;                         // NBS
    float* qb   = h    + NBS;                         // NBS
    float* kb   = qb   + NBS;                         // NBS
    float* vb   = kb   + NBS;                         // NBS
    float* attn = vb   + NBS;                         // NBS
    float* act  = attn + NBS;                         // MROWS*HE
    float* tb = kb;   // wo-proj tmp (k free after attention)
    float* xb = vb;   // post-attn residual x (v free after attention)
    float* yb = qb;   // MLP/MoE accumulator (q free after attention)

    // 1) adaLN modulation
    hd455_silu_small<<<dim3((BATCH * DIM) / 256), 256, 0, stream>>>(adaln, sa, BATCH * DIM);
    hd455_adaln_gemm<<<dim3((BATCH * D6) / 256), 256, 0, stream>>>(sa, adaw, adab, mod);

    // 2) h = LN(x)*(1+sc_msa)+sh_msa
    hd455_ln_mod<<<dim3(MROWS), 256, 0, stream>>>(x_in, mod, 0, DIM, h);

    // 3) QKV projections (WMMA GEMMs)
    dim3 gP(DIM / 128, MROWS / 128);
    hd455_gemm<false, false><<<gP, 256, 0, stream>>>(h, wq, qb, MROWS, DIM, DIM, nullptr, 0);
    hd455_gemm<false, false><<<gP, 256, 0, stream>>>(h, wk, kb, MROWS, DIM, DIM, nullptr, 0);
    hd455_gemm<false, false><<<gP, 256, 0, stream>>>(h, wv, vb, MROWS, DIM, DIM, nullptr, 0);

    // 4) RMS norms + RoPE
    hd455_rms<<<dim3(MROWS), 256, 0, stream>>>(qb, qrms);
    hd455_rms<<<dim3(MROWS), 256, 0, stream>>>(kb, krms);
    hd455_rope<<<dim3((BATCH * SEQ * HEADS * (HD / 2)) / 256), 256, 0, stream>>>(qb, kb, rope);

    // 5) flash attention
    hd455_attn<<<dim3(SEQ / 128, BATCH * HEADS), 256, 0, stream>>>(qb, kb, vb, attn);

    // 6) x = x + g_msa * (attn @ wo)
    hd455_gemm<false, false><<<gP, 256, 0, stream>>>(attn, wo, tb, MROWS, DIM, DIM, nullptr, 0);
    hd455_residual<<<dim3((int)(NBS / 256)), 256, 0, stream>>>(x_in, tb, mod, 2 * DIM, xb, (int)NBS);

    // 7) h = LN(x)*(1+sc_mlp)+sh_mlp
    hd455_ln_mod<<<dim3(MROWS), 256, 0, stream>>>(xb, mod, 3 * DIM, 4 * DIM, h);

    // 8) gating
    hd455_gate<<<dim3(MROWS), 64, 0, stream>>>(h, gw, cw);

    // 9) MoE experts: act = silu(h@ew1)*(h@ew3); y (+)= cw[:,e] * (act@ew2)
    dim3 gDualE(HE / 128, MROWS / 64);
    for (int e = 0; e < NE; ++e) {
        hd455_gemm_dual_silu<<<gDualE, 256, 0, stream>>>(
            h, ew1 + (size_t)e * DIM * HE, ew3 + (size_t)e * DIM * HE, act, MROWS, HE, DIM);
        if (e == 0)
            hd455_gemm<true, false><<<gP, 256, 0, stream>>>(
                act, ew2 + (size_t)e * HE * DIM, yb, MROWS, DIM, HE, cw + e, NE);
        else
            hd455_gemm<true, true><<<gP, 256, 0, stream>>>(
                act, ew2 + (size_t)e * HE * DIM, yb, MROWS, DIM, HE, cw + e, NE);
    }

    // 10) shared MLP: y += (silu(h@sw1)*(h@sw3)) @ sw2
    dim3 gDualS(HS / 128, MROWS / 64);
    hd455_gemm_dual_silu<<<gDualS, 256, 0, stream>>>(h, sw1, sw3, act, MROWS, HS, DIM);
    hd455_gemm<false, true><<<gP, 256, 0, stream>>>(act, sw2, yb, MROWS, DIM, HS, nullptr, 0);

    // 11) out = x + g_mlp * y
    hd455_residual<<<dim3((int)(NBS / 256)), 256, 0, stream>>>(xb, yb, mod, 5 * DIM, out, (int)NBS);
}
